// MultiHeadAttention_7653631722147
// MI455X (gfx1250) — matrix operand order repaired and checked by reading, not compiled, not run on hardware
//
#include <hip/hip_runtime.h>
#include <hip/hip_bf16.h>
#include <math.h>

// ---------------------------------------------------------------------------
// MI455X (gfx1250) multi-head attention forward.
// GEMMs: v_wmma_f32_16x16x32_f16 (f16 A/B, f32 accumulate), software-pipelined
// K-loops; A-tiles staged to LDS via async global->LDS copies where available.
// ---------------------------------------------------------------------------

typedef __attribute__((ext_vector_type(16))) _Float16 v16h;
typedef __attribute__((ext_vector_type(8)))  _Float16 v8h;
typedef __attribute__((ext_vector_type(8)))  float    v8f;
typedef __attribute__((ext_vector_type(4)))  int      v4i;

#define EMBED 512
#define HEADS 8
#define HDIM  64
#define BATCH 2
#define SEQ   2048
#define NTOK  (BATCH * SEQ)            // 4096
#define OUT_ELEMS ((size_t)BATCH * SEQ * EMBED)        // 2,097,152 f32

// ---- workspace layout (f16 element offsets; total ~23 MB) ------------------
#define WS_XH   ((size_t)0)                         // [NTOK][EMBED]
#define WS_WQT  ((size_t)2097152)                   // [EMBED][EMBED] (W^T)
#define WS_WKT  ((size_t)2359296)
#define WS_WVT  ((size_t)2621440)
#define WS_WOT  ((size_t)2883584)
#define WS_QH   ((size_t)3145728)                   // [B*H][S][Dh]
#define WS_KH   ((size_t)5242880)                   // [B*H][S][Dh]
#define WS_VT   ((size_t)7340032)                   // [B*H][Dh][S]  (transposed)
#define WS_ATTN ((size_t)9437184)                   // [NTOK][EMBED] concat

// ---- async global -> LDS staging (gfx1250) --------------------------------
#if __has_builtin(__builtin_amdgcn_global_load_async_to_lds_b128) && \
    __has_builtin(__builtin_amdgcn_s_wait_asynccnt)
#define HAVE_ASYNC_LDS 1
#endif

// Copy `chunks16` 16-byte chunks from gsrc to LDS dst, whole workgroup (256T).
__device__ __forceinline__ void stage_to_lds_16B(_Float16* sdst,
                                                 const _Float16* gsrc,
                                                 int chunks16) {
  const int tid = threadIdx.x;
#ifdef HAVE_ASYNC_LDS
  for (int i = tid; i < chunks16; i += 256) {
    __builtin_amdgcn_global_load_async_to_lds_b128(
        (v4i*)(gsrc + (size_t)i * 8),
        (v4i*)(sdst + (size_t)i * 8),
        0, 0);
  }
  __builtin_amdgcn_s_wait_asynccnt(0);
#else
  for (int i = tid; i < chunks16; i += 256) {
    *(v8h*)(sdst + (size_t)i * 8) = *(const v8h*)(gsrc + (size_t)i * 8);
  }
#endif
  __syncthreads();
}

// ---------------------------------------------------------------------------
// WMMA helpers (A and B both use contiguous k order, k = 16*(lane/16) + s)
// ---------------------------------------------------------------------------
__device__ __forceinline__ v8f wmma16(v16h a, v16h b, v8f c) {
  return __builtin_amdgcn_wmma_f32_16x16x32_f16(
      /*neg_a=*/false, a, /*neg_b=*/false, b,
      /*c_mod=*/(short)0, c, /*reuse_a=*/false, /*reuse_b=*/false);
}

// A fragment (16x32) from an f16 row pointer (global or LDS).
__device__ __forceinline__ v16h make_a_f16(const _Float16* rowk) {
  const int lane = threadIdx.x & 31;
  const int lo = (lane & 16) ? 16 : 0;
  v8h x0 = *(const v8h*)(rowk + lo);
  v8h x1 = *(const v8h*)(rowk + lo + 8);
  v16h a;
#pragma unroll
  for (int i = 0; i < 8; ++i) { a[i] = x0[i]; a[8 + i] = x1[i]; }
  return a;
}

// A fragment built from an f32 source row (converted in-register).
__device__ __forceinline__ v16h make_a_f32(const float* rowk) {
  const int lane = threadIdx.x & 31;
  const int lo = (lane & 16) ? 16 : 0;
  v8f x0 = *(const v8f*)(rowk + lo);
  v8f x1 = *(const v8f*)(rowk + lo + 8);
  v16h a;
#pragma unroll
  for (int i = 0; i < 8; ++i) { a[i] = (_Float16)x0[i]; a[8 + i] = (_Float16)x1[i]; }
  return a;
}

// B fragment (32x16): BT[n][k] column-major B; 32B contiguous per lane.
__device__ __forceinline__ v16h make_b(const _Float16* BT, int ldb, int ncol, int k0) {
  const int lane = threadIdx.x & 31;
  const int hi = (lane & 16) ? 16 : 0;
  return *(const v16h*)(BT + (size_t)(ncol + (lane & 15)) * ldb + k0 + hi);
}

// ---------------------------------------------------------------------------
// Kernel 1a: f32 -> f16 convert of x
// ---------------------------------------------------------------------------
__global__ void mha_cvt_x(const float* __restrict__ x, _Float16* __restrict__ xh) {
  size_t i = (size_t)blockIdx.x * 256 + threadIdx.x;   // 2,097,152 elems
  xh[i] = (_Float16)x[i];
}

// Kernel 1b: transpose+convert all four weight matrices (grid.y selects)
__global__ void mha_cvt_wT(const float* __restrict__ w0, const float* __restrict__ w1,
                           const float* __restrict__ w2, const float* __restrict__ w3,
                           _Float16* __restrict__ o0, _Float16* __restrict__ o1,
                           _Float16* __restrict__ o2, _Float16* __restrict__ o3) {
  const float* src = (blockIdx.y == 0) ? w0 : (blockIdx.y == 1) ? w1
                   : (blockIdx.y == 2) ? w2 : w3;
  _Float16* dst    = (blockIdx.y == 0) ? o0 : (blockIdx.y == 1) ? o1
                   : (blockIdx.y == 2) ? o2 : o3;
  int i = blockIdx.x * 256 + threadIdx.x;              // 262,144 elems
  int n = i >> 9, k = i & 511;
  dst[i] = (_Float16)src[(size_t)k * EMBED + n];       // dst[n][k] = src[k][n]
}

// ---------------------------------------------------------------------------
// Kernel 2: fused QKV projection.  grid.y in {0=Q,1=K,2=V}.
// All 8 waves of a block share one 16-row A tile -> staged to LDS (async).
// Each wave owns a 64-wide column strip; K-loop software-pipelined on B.
// ---------------------------------------------------------------------------
__global__ void __launch_bounds__(256)
mha_qkv(const _Float16* __restrict__ xh,
        const _Float16* __restrict__ WqT, const _Float16* __restrict__ WkT,
        const _Float16* __restrict__ WvT,
        const float* __restrict__ bq, const float* __restrict__ bk,
        const float* __restrict__ bv,
        _Float16* __restrict__ qh, _Float16* __restrict__ kh,
        _Float16* __restrict__ vT) {
  __shared__ _Float16 sA[16 * EMBED];          // 16 KB A tile

  const int which = blockIdx.y;
  const _Float16* WT = (which == 0) ? WqT : (which == 1) ? WkT : WvT;
  const float*   bias = (which == 0) ? bq  : (which == 1) ? bk  : bv;

  const int wave  = threadIdx.x >> 5;
  const int lane  = threadIdx.x & 31;
  const int m0 = blockIdx.x << 4;              // shared token tile
  const int n0 = wave << 6;                    // per-wave 64-wide strip

  stage_to_lds_16B(sA, xh + (size_t)m0 * EMBED, 16 * EMBED / 8);

  const _Float16* arow = sA + (size_t)(lane & 15) * EMBED;

  v8f c0 = {}, c1 = {}, c2 = {}, c3 = {};
  v16h b0 = make_b(WT, EMBED, n0,      0);
  v16h b1 = make_b(WT, EMBED, n0 + 16, 0);
  v16h b2 = make_b(WT, EMBED, n0 + 32, 0);
  v16h b3 = make_b(WT, EMBED, n0 + 48, 0);
  for (int k0 = 32; k0 < EMBED; k0 += 32) {
    v16h nb0 = make_b(WT, EMBED, n0,      k0);
    v16h nb1 = make_b(WT, EMBED, n0 + 16, k0);
    v16h nb2 = make_b(WT, EMBED, n0 + 32, k0);
    v16h nb3 = make_b(WT, EMBED, n0 + 48, k0);
    v16h a = make_a_f16(arow + k0 - 32);
    c0 = wmma16(a, b0, c0);
    c1 = wmma16(a, b1, c1);
    c2 = wmma16(a, b2, c2);
    c3 = wmma16(a, b3, c3);
    b0 = nb0; b1 = nb1; b2 = nb2; b3 = nb3;
  }
  {
    v16h a = make_a_f16(arow + EMBED - 32);
    c0 = wmma16(a, b0, c0);
    c1 = wmma16(a, b1, c1);
    c2 = wmma16(a, b2, c2);
    c3 = wmma16(a, b3, c3);
  }

  const int hi8 = (lane & 16) ? 8 : 0;
  v8f acc[4] = {c0, c1, c2, c3};
#pragma unroll
  for (int j = 0; j < 4; ++j) {
#pragma unroll
    for (int r = 0; r < 8; ++r) {
      int t = m0 + r + hi8;                     // token index
      int e = n0 + j * 16 + (lane & 15);        // embed column
      float val = acc[j][r] + bias[e];
      int bb = t >> 11, ss = t & 2047;
      int hh = e >> 6,  dd = e & 63;
      size_t bh = (size_t)bb * HEADS + hh;
      if (which == 2) {
        vT[(bh * HDIM + dd) * SEQ + ss] = (_Float16)val;        // V transposed
      } else {
        _Float16* dst = (which == 0) ? qh : kh;
        dst[(bh * SEQ + ss) * HDIM + dd] = (_Float16)val;
      }
    }
  }
}

// ---------------------------------------------------------------------------
// Kernel 3: scores = (Q K^T) * 1/sqrt(Dh), f32 into weights region of d_out.
// grid.y = b*H+h (16).  Per wave: 16x64 strip of the [S x S] score matrix.
// ---------------------------------------------------------------------------
__global__ void __launch_bounds__(256)
mha_scores(const _Float16* __restrict__ qh, const _Float16* __restrict__ kh,
           float* __restrict__ wts) {
  const size_t bh = blockIdx.y;
  const int wave = threadIdx.x >> 5;
  const int lane = threadIdx.x & 31;
  const int strip = blockIdx.x * 8 + wave;   // 4096 strips = 128 mtiles * 32
  const int m0 = (strip >> 5) << 4;          // query rows
  const int n0 = (strip & 31) << 6;          // key columns (64-wide)

  const _Float16* qbase = qh + bh * (size_t)SEQ * HDIM;
  const _Float16* kbase = kh + bh * (size_t)SEQ * HDIM;
  const _Float16* arow  = qbase + (size_t)(m0 + (lane & 15)) * HDIM;

  v8f c0 = {}, c1 = {}, c2 = {}, c3 = {};
#pragma unroll
  for (int k0 = 0; k0 < HDIM; k0 += 32) {
    v16h t0 = make_b(kbase, HDIM, n0,      k0);
    v16h t1 = make_b(kbase, HDIM, n0 + 16, k0);
    v16h t2 = make_b(kbase, HDIM, n0 + 32, k0);
    v16h t3 = make_b(kbase, HDIM, n0 + 48, k0);
    v16h a  = make_a_f16(arow + k0);
    c0 = wmma16(a, t0, c0);
    c1 = wmma16(a, t1, c1);
    c2 = wmma16(a, t2, c2);
    c3 = wmma16(a, t3, c3);
  }

  const int hi8 = (lane & 16) ? 8 : 0;
  float* wbase = wts + bh * (size_t)SEQ * SEQ;
  v8f acc[4] = {c0, c1, c2, c3};
#pragma unroll
  for (int j = 0; j < 4; ++j)
#pragma unroll
    for (int r = 0; r < 8; ++r) {
      int row = m0 + r + hi8;
      int col = n0 + j * 16 + (lane & 15);
      wbase[(size_t)row * SEQ + col] = acc[j][r] * 0.125f;   // 1/sqrt(64)
    }
}

// ---------------------------------------------------------------------------
// Kernel 4: in-place row softmax over the weights region (row length 2048).
// One 256-thread workgroup per row; LDS tree reductions.
// ---------------------------------------------------------------------------
__global__ void __launch_bounds__(256)
mha_softmax(float* __restrict__ wts) {
  __shared__ float red[256];
  float* p = wts + (size_t)blockIdx.x * SEQ;
  const int t = threadIdx.x;

  float v[8];
  float mx = -INFINITY;
#pragma unroll
  for (int j = 0; j < 8; ++j) { v[j] = p[t + j * 256]; mx = fmaxf(mx, v[j]); }
  red[t] = mx;
  __syncthreads();
  for (int s = 128; s > 0; s >>= 1) {
    if (t < s) red[t] = fmaxf(red[t], red[t + s]);
    __syncthreads();
  }
  mx = red[0];
  __syncthreads();

  float sum = 0.f;
#pragma unroll
  for (int j = 0; j < 8; ++j) { v[j] = __expf(v[j] - mx); sum += v[j]; }
  red[t] = sum;
  __syncthreads();
  for (int s = 128; s > 0; s >>= 1) {
    if (t < s) red[t] += red[t + s];
    __syncthreads();
  }
  float inv = 1.0f / red[0];
#pragma unroll
  for (int j = 0; j < 8; ++j) p[t + j * 256] = v[j] * inv;
}

// ---------------------------------------------------------------------------
// Kernel 5: attn = softmax(W) @ V.  A built from f32 weights (cvt in-reg),
// B from pre-transposed V.  Per wave: 16 rows x full Dh (64); K=S=2048,
// software-pipelined (streaming 268 MB read -> prefetch ahead).
// ---------------------------------------------------------------------------
__global__ void __launch_bounds__(256)
mha_attnv(const float* __restrict__ wts, const _Float16* __restrict__ vT,
          _Float16* __restrict__ attnh) {
  const size_t bh = blockIdx.y;
  const int wave = threadIdx.x >> 5;
  const int lane = threadIdx.x & 31;
  const int strip = blockIdx.x * 8 + wave;   // 128 strips/head
  const int m0 = strip << 4;

  const float*    arow  = wts + bh * (size_t)SEQ * SEQ
                              + (size_t)(m0 + (lane & 15)) * SEQ;
  const _Float16* vbase = vT + bh * (size_t)HDIM * SEQ;

  v8f c0 = {}, c1 = {}, c2 = {}, c3 = {};
  v16h a  = make_a_f32(arow);
  v16h b0 = make_b(vbase, SEQ, 0,  0);
  v16h b1 = make_b(vbase, SEQ, 16, 0);
  v16h b2 = make_b(vbase, SEQ, 32, 0);
  v16h b3 = make_b(vbase, SEQ, 48, 0);
  for (int k0 = 32; k0 < SEQ; k0 += 32) {
    __builtin_prefetch(arow + k0 + 224, 0, 3);
    v16h na  = make_a_f32(arow + k0);
    v16h nb0 = make_b(vbase, SEQ, 0,  k0);
    v16h nb1 = make_b(vbase, SEQ, 16, k0);
    v16h nb2 = make_b(vbase, SEQ, 32, k0);
    v16h nb3 = make_b(vbase, SEQ, 48, k0);
    c0 = wmma16(a, b0, c0);
    c1 = wmma16(a, b1, c1);
    c2 = wmma16(a, b2, c2);
    c3 = wmma16(a, b3, c3);
    a = na; b0 = nb0; b1 = nb1; b2 = nb2; b3 = nb3;
  }
  c0 = wmma16(a, b0, c0);
  c1 = wmma16(a, b1, c1);
  c2 = wmma16(a, b2, c2);
  c3 = wmma16(a, b3, c3);

  const int hi8 = (lane & 16) ? 8 : 0;
  const int b = (int)(bh >> 3), h = (int)(bh & 7);
  v8f acc[4] = {c0, c1, c2, c3};
#pragma unroll
  for (int j = 0; j < 4; ++j)
#pragma unroll
    for (int r = 0; r < 8; ++r) {
      int s = m0 + r + hi8;
      int d = j * 16 + (lane & 15);
      size_t tok = (size_t)b * SEQ + s;
      attnh[tok * EMBED + h * HDIM + d] = (_Float16)acc[j][r];
    }
}

// ---------------------------------------------------------------------------
// Kernel 6: out = attn_concat @ Wo + bo  (f32 to d_out).
// Same structure as QKV: shared 16-row A tile staged to LDS, pipelined B.
// ---------------------------------------------------------------------------
__global__ void __launch_bounds__(256)
mha_outproj(const _Float16* __restrict__ attnh, const _Float16* __restrict__ WoT,
            const float* __restrict__ bo, float* __restrict__ out) {
  __shared__ _Float16 sA[16 * EMBED];

  const int wave = threadIdx.x >> 5;
  const int lane = threadIdx.x & 31;
  const int m0 = blockIdx.x << 4;
  const int n0 = wave << 6;

  stage_to_lds_16B(sA, attnh + (size_t)m0 * EMBED, 16 * EMBED / 8);

  const _Float16* arow = sA + (size_t)(lane & 15) * EMBED;

  v8f c0 = {}, c1 = {}, c2 = {}, c3 = {};
  v16h b0 = make_b(WoT, EMBED, n0,      0);
  v16h b1 = make_b(WoT, EMBED, n0 + 16, 0);
  v16h b2 = make_b(WoT, EMBED, n0 + 32, 0);
  v16h b3 = make_b(WoT, EMBED, n0 + 48, 0);
  for (int k0 = 32; k0 < EMBED; k0 += 32) {
    v16h nb0 = make_b(WoT, EMBED, n0,      k0);
    v16h nb1 = make_b(WoT, EMBED, n0 + 16, k0);
    v16h nb2 = make_b(WoT, EMBED, n0 + 32, k0);
    v16h nb3 = make_b(WoT, EMBED, n0 + 48, k0);
    v16h a = make_a_f16(arow + k0 - 32);
    c0 = wmma16(a, b0, c0);
    c1 = wmma16(a, b1, c1);
    c2 = wmma16(a, b2, c2);
    c3 = wmma16(a, b3, c3);
    b0 = nb0; b1 = nb1; b2 = nb2; b3 = nb3;
  }
  {
    v16h a = make_a_f16(arow + EMBED - 32);
    c0 = wmma16(a, b0, c0);
    c1 = wmma16(a, b1, c1);
    c2 = wmma16(a, b2, c2);
    c3 = wmma16(a, b3, c3);
  }

  const int hi8 = (lane & 16) ? 8 : 0;
  v8f acc[4] = {c0, c1, c2, c3};
#pragma unroll
  for (int j = 0; j < 4; ++j)
#pragma unroll
    for (int r = 0; r < 8; ++r) {
      int t = m0 + r + hi8;
      int e = n0 + j * 16 + (lane & 15);
      out[(size_t)t * EMBED + e] = acc[j][r] + bo[e];
    }
}

// ---------------------------------------------------------------------------
extern "C" void kernel_launch(void* const* d_in, const int* in_sizes, int n_in,
                              void* d_out, int out_size, void* d_ws, size_t ws_size,
                              hipStream_t stream) {
  const float* x  = (const float*)d_in[0];
  const float* Wq = (const float*)d_in[1];
  const float* bq = (const float*)d_in[2];
  const float* Wk = (const float*)d_in[3];
  const float* bk = (const float*)d_in[4];
  const float* Wv = (const float*)d_in[5];
  const float* bv = (const float*)d_in[6];
  const float* Wo = (const float*)d_in[7];
  const float* bo = (const float*)d_in[8];

  float* out = (float*)d_out;
  float* wts = out + OUT_ELEMS;                 // [B*H][S][S]

  _Float16* ws   = (_Float16*)d_ws;
  _Float16* xh   = ws + WS_XH;
  _Float16* WqT  = ws + WS_WQT;
  _Float16* WkT  = ws + WS_WKT;
  _Float16* WvT  = ws + WS_WVT;
  _Float16* WoT  = ws + WS_WOT;
  _Float16* qh   = ws + WS_QH;
  _Float16* kh   = ws + WS_KH;
  _Float16* vT   = ws + WS_VT;
  _Float16* atth = ws + WS_ATTN;

  // 1) precision conversion / weight transposes
  mha_cvt_x<<<dim3(8192), dim3(256), 0, stream>>>(x, xh);
  mha_cvt_wT<<<dim3(1024, 4), dim3(256), 0, stream>>>(Wq, Wk, Wv, Wo,
                                                      WqT, WkT, WvT, WoT);
  // 2) QKV projections (grid.y selects Q/K/V)
  mha_qkv<<<dim3(256, 3), dim3(256), 0, stream>>>(xh, WqT, WkT, WvT,
                                                  bq, bk, bv, qh, kh, vT);
  // 3) scores -> weights region of d_out
  mha_scores<<<dim3(512, BATCH * HEADS), dim3(256), 0, stream>>>(qh, kh, wts);
  // 4) softmax rows in place
  mha_softmax<<<dim3(BATCH * HEADS * SEQ), dim3(256), 0, stream>>>(wts);
  // 5) attn = W @ V
  mha_attnv<<<dim3(16, BATCH * HEADS), dim3(256), 0, stream>>>(wts, vT, atth);
  // 6) output projection
  mha_outproj<<<dim3(256), dim3(256), 0, stream>>>(atth, WoT, bo, out);
}
